// MultiHeadAttention_16741782520451
// MI455X (gfx1250) — matrix operand order repaired and checked by reading, not compiled, not run on hardware
//
#include <hip/hip_runtime.h>
#include <hip/hip_bf16.h>

// ---------------------------------------------------------------------------
// MHA forward for MI455X (gfx1250): f16 WMMA (v_wmma_f32_16x16x32_f16) with
// f32 accumulation everywhere; flash-attention with transposed-score trick so
// P feeds the P@V WMMA lane-locally (no cross-lane transpose).
// ---------------------------------------------------------------------------

typedef __attribute__((ext_vector_type(16))) _Float16 v16h;
typedef __attribute__((ext_vector_type(8)))  _Float16 h8;
typedef __attribute__((ext_vector_type(4)))  _Float16 h4;
typedef __attribute__((ext_vector_type(8)))  float    v8f;

#define HEADS  16
#define DMODEL 1024
#define HDIM   64
#define BATCH  2
#define SEQ    2048
#define MTOT   (BATCH * SEQ)     // 4096 token rows
#define NEGV   (-1.0e15f)

static __device__ __forceinline__ v8f vzero8() {
  v8f z;
#pragma unroll
  for (int i = 0; i < 8; ++i) z[i] = 0.0f;
  return z;
}

// A operand (16x32 f16, M=rows): lane holds row (lane&15); halves:
//   elems 0-7  -> K = k0 + 8*(lane>>4) + e
//   elems 8-15 -> K = k0 + 8*(lane>>4) + 16 + (e-8)
static __device__ __forceinline__ v16h load_afrag(const _Float16* __restrict__ row, int lg) {
  union { v16h v; h8 h[2]; } u;
  u.h[0] = *(const h8*)(row + lg * 8);
  u.h[1] = *(const h8*)(row + lg * 8 + 16);
  return u.v;
}

// B operand (32x16 f16, N=cols): lane holds col (lane&15); elems 0-7 -> K = k0 + 8*(lane>>4) + e,
// elems 8-15 -> K = k0 + 8*(lane>>4) + 16 + (e-8). p points at k0 + 8*(lane>>4) in a row of the [N,K] matrix.
static __device__ __forceinline__ v16h load_bfrag(const _Float16* __restrict__ p) {
  union { v16h v; h8 h[2]; } u;
  u.h[0] = *(const h8*)(p);
  u.h[1] = *(const h8*)(p + 16);
  return u.v;
}

static __device__ __forceinline__ v8f wmma_f16(v16h a, v16h b, v8f c) {
  return __builtin_amdgcn_wmma_f32_16x16x32_f16(false, a, false, b, (short)0, c, false, false);
}

// ---------------------------------------------------------------------------
// f32 -> f16 conversion (vectorized x4)
// ---------------------------------------------------------------------------
__global__ void cvt_f32_to_f16(const float* __restrict__ x, _Float16* __restrict__ y, int n4) {
  int i = blockIdx.x * blockDim.x + threadIdx.x;
  if (i < n4) {
    float4 v = reinterpret_cast<const float4*>(x)[i];
    h4 o;
    o[0] = (_Float16)v.x; o[1] = (_Float16)v.y;
    o[2] = (_Float16)v.z; o[3] = (_Float16)v.w;
    reinterpret_cast<h4*>(y)[i] = o;
  }
}

// ---------------------------------------------------------------------------
// Core 32(M)x64(N) GEMM tile per wave: acc = X[32,1024] * W^T (W is [N,K] row-major)
// ---------------------------------------------------------------------------
static __device__ __forceinline__ void gemm_tile_32x64(
    const _Float16* __restrict__ X, const _Float16* __restrict__ W,
    int mbase, int nbase, int lr, int lg, v8f acc[2][4]) {
#pragma unroll
  for (int i = 0; i < 2; ++i)
#pragma unroll
    for (int j = 0; j < 4; ++j) acc[i][j] = vzero8();

  const _Float16* xr0 = X + (size_t)(mbase + lr) * DMODEL;
  const _Float16* xr1 = X + (size_t)(mbase + 16 + lr) * DMODEL;
  for (int k0 = 0; k0 < DMODEL; k0 += 32) {
    v16h a0 = load_afrag(xr0 + k0, lg);
    v16h a1 = load_afrag(xr1 + k0, lg);
    v16h b[4];
#pragma unroll
    for (int j = 0; j < 4; ++j)
      b[j] = load_bfrag(W + (size_t)(nbase + j * 16 + lr) * DMODEL + k0 + lg * 8);
#pragma unroll
    for (int j = 0; j < 4; ++j) {
      acc[0][j] = wmma_f16(a0, b[j], acc[0][j]);
      acc[1][j] = wmma_f16(a1, b[j], acc[1][j]);
    }
  }
}

// ---------------------------------------------------------------------------
// Fused QKV projection: z=0 -> Q (scaled 1/8, [B,H,S,64]); z=1 -> K ([B,H,S,64]);
// z=2 -> V transposed ([B,H,64,S], packed b128 stores)
// ---------------------------------------------------------------------------
__global__ void proj_qkv(const _Float16* __restrict__ xq, const _Float16* __restrict__ xk,
                         const _Float16* __restrict__ xv,
                         const _Float16* __restrict__ wq, const _Float16* __restrict__ wk,
                         const _Float16* __restrict__ wv,
                         const float* __restrict__ bq, const float* __restrict__ bk,
                         const float* __restrict__ bv,
                         _Float16* __restrict__ Qh, _Float16* __restrict__ Kh,
                         _Float16* __restrict__ Vt) {
  const int z = blockIdx.z;
  const _Float16* X = (z == 0) ? xq : (z == 1) ? xk : xv;
  const _Float16* W = (z == 0) ? wq : (z == 1) ? wk : wv;
  const float* bias = (z == 0) ? bq : (z == 1) ? bk : bv;
  _Float16* dst     = (z == 0) ? Qh : (z == 1) ? Kh : Vt;
  const float scale = (z == 0) ? 0.125f : 1.0f;   // 1/sqrt(HDIM) folded into Q

  const int wid  = blockIdx.x * (blockDim.x >> 5) + (threadIdx.x >> 5);
  const int lane = threadIdx.x & 31;
  const int lg = lane >> 4, lr = lane & 15;
  const int tm = wid >> 4;          // 128 M-tiles of 32
  const int tn = wid & 15;          // 16 N-tiles of 64
  const int mbase = tm * 32, nbase = tn * 64;

  v8f acc[2][4];
  gemm_tile_32x64(X, W, mbase, nbase, lr, lg, acc);

#pragma unroll
  for (int j = 0; j < 4; ++j) {
    const int n = nbase + j * 16 + lr;       // output feature (lane-fixed column)
    const float bn = bias[n];
    const int h = n >> 6, d = n & 63;
#pragma unroll
    for (int i = 0; i < 2; ++i) {
      const int m0 = mbase + i * 16 + lg * 8;  // first of 8 consecutive token rows
      const int b = m0 >> 11;                  // /SEQ
      const int s0 = m0 & (SEQ - 1);
      if (z == 2) {
        // V transposed [B,H,64,S]: 8 rows -> 8 contiguous halves along S
        h8 pk;
#pragma unroll
        for (int r = 0; r < 8; ++r) pk[r] = (_Float16)(acc[i][j][r] + bn);
        *(h8*)(Vt + ((size_t)(b * HEADS + h) * HDIM + d) * SEQ + s0) = pk;
      } else {
#pragma unroll
        for (int r = 0; r < 8; ++r)
          dst[((size_t)(b * HEADS + h) * SEQ + (s0 + r)) * HDIM + d] =
              (_Float16)((acc[i][j][r] + bn) * scale);
      }
    }
  }
}

// ---------------------------------------------------------------------------
// Flash attention: one wave per (b,h,16-query tile); 32 keys per iteration.
// S^T = K_tile @ Q^T so exp'd scores map lane-locally into P's A-fragment.
// ---------------------------------------------------------------------------
__global__ void attn_flash(const _Float16* __restrict__ Qh, const _Float16* __restrict__ Kh,
                           const _Float16* __restrict__ Vt, _Float16* __restrict__ ctx) {
  const int wid  = blockIdx.x * (blockDim.x >> 5) + (threadIdx.x >> 5);
  const int lane = threadIdx.x & 31;
  const int lg = lane >> 4, lr = lane & 15;
  const int QT = SEQ / 16;                  // 128 query tiles per head
  const int qt = wid & (QT - 1);
  const int bh = wid / QT;                  // 0..B*HEADS-1
  const int qbase = qt * 16;

  const _Float16* Q = Qh + (size_t)bh * SEQ * HDIM;
  const _Float16* K = Kh + (size_t)bh * SEQ * HDIM;
  const _Float16* V = Vt + (size_t)bh * HDIM * SEQ;

  // Q^T as B operand for S^T = K @ Q^T; lane column = query lr. Held all loop.
  v16h qb[2];
#pragma unroll
  for (int s = 0; s < 2; ++s)
    qb[s] = load_bfrag(Q + (size_t)(qbase + lr) * HDIM + s * 32 + lg * 8);

  v8f oacc[4];
#pragma unroll
  for (int j = 0; j < 4; ++j) oacc[j] = vzero8();
  float mrow = -3.0e38f;   // running max for this lane's query (lr)
  float lsum = 0.0f;       // running denom
  const int q_idx = qbase + lr;

  const int nblk = (qbase >> 5) + 1;        // causal: keys 0 .. qbase+15
  for (int kb = 0; kb < nblk; ++kb) {
    const int k32 = kb * 32;

    // --- transposed scores for 32 keys: st[t] covers keys [k32+t*16, +16) ---
    v8f st[2];
#pragma unroll
    for (int t = 0; t < 2; ++t) {
      st[t] = vzero8();
#pragma unroll
      for (int s = 0; s < 2; ++s) {
        v16h ka = load_afrag(K + (size_t)(k32 + t * 16 + lr) * HDIM + s * 32, lg);
        st[t] = wmma_f16(ka, qb[s], st[t]);
      }
    }

    // --- causal mask + online softmax (row = query = lr, keys split lg/lg^1) ---
    float sm[16];
    float bm = -3.0e38f;
#pragma unroll
    for (int t = 0; t < 2; ++t)
#pragma unroll
      for (int r = 0; r < 8; ++r) {
        const int key = k32 + t * 16 + lg * 8 + r;
        float sv = st[t][r];
        if (key > q_idx) sv += NEGV;        // matches reference scores + mask*NEG
        sm[t * 8 + r] = sv;
        bm = fmaxf(bm, sv);
      }
    bm = fmaxf(bm, __shfl_xor(bm, 16, 32)); // other 16-key half of this query
    const float mnew  = fmaxf(mrow, bm);
    const float alpha = __expf(mrow - mnew);

    float rs = 0.0f;
    v16h pa;                                 // P as A-fragment (lane-local!)
#pragma unroll
    for (int e = 0; e < 16; ++e) {
      const float ex = __expf(sm[e] - mnew);
      rs += ex;
      pa[e] = (_Float16)ex;
    }
    rs += __shfl_xor(rs, 16, 32);
    lsum = lsum * alpha + rs;
    mrow = mnew;

    // broadcast alpha from query-lane to output-row layout, rescale acc
    float arow[8];
#pragma unroll
    for (int r = 0; r < 8; ++r) arow[r] = __shfl(alpha, lg * 8 + r, 32);
#pragma unroll
    for (int j = 0; j < 4; ++j)
#pragma unroll
      for (int r = 0; r < 8; ++r) oacc[j][r] *= arow[r];

    // --- ctx += P(16x32) @ V(32x64): B from transposed V is contiguous ---
#pragma unroll
    for (int j = 0; j < 4; ++j) {
      v16h vb = load_bfrag(V + (size_t)(j * 16 + lr) * SEQ + k32 + lg * 8);
      oacc[j] = wmma_f16(pa, vb, oacc[j]);
    }
  }

  // --- normalize and store ctx (merged heads, [B,S,DMODEL] f16) ---
  const float il = 1.0f / lsum;
  float ilr[8];
#pragma unroll
  for (int r = 0; r < 8; ++r) ilr[r] = __shfl(il, lg * 8 + r, 32);

  const int b = bh / HEADS, h = bh % HEADS;
#pragma unroll
  for (int j = 0; j < 4; ++j) {
    const int feat = h * HDIM + j * 16 + lr;
#pragma unroll
    for (int r = 0; r < 8; ++r) {
      const int s = qbase + lg * 8 + r;
      ctx[((size_t)b * SEQ + s) * DMODEL + feat] = (_Float16)(oacc[j][r] * ilr[r]);
    }
  }
}

// ---------------------------------------------------------------------------
// Output projection: out = ctx @ Wo^T + bo (f32 store)
// ---------------------------------------------------------------------------
__global__ void proj_out(const _Float16* __restrict__ X, const _Float16* __restrict__ Wo,
                         const float* __restrict__ bo, float* __restrict__ out) {
  const int wid  = blockIdx.x * (blockDim.x >> 5) + (threadIdx.x >> 5);
  const int lane = threadIdx.x & 31;
  const int lg = lane >> 4, lr = lane & 15;
  const int tm = wid >> 4, tn = wid & 15;
  const int mbase = tm * 32, nbase = tn * 64;

  v8f acc[2][4];
  gemm_tile_32x64(X, Wo, mbase, nbase, lr, lg, acc);

#pragma unroll
  for (int j = 0; j < 4; ++j) {
    const int n = nbase + j * 16 + lr;
    const float bn = bo[n];
#pragma unroll
    for (int i = 0; i < 2; ++i) {
      const int m0 = mbase + i * 16 + lg * 8;
#pragma unroll
      for (int r = 0; r < 8; ++r)
        out[(size_t)(m0 + r) * DMODEL + n] = acc[i][j][r] + bn;
    }
  }
}

// ---------------------------------------------------------------------------
extern "C" void kernel_launch(void* const* d_in, const int* in_sizes, int n_in,
                              void* d_out, int out_size, void* d_ws, size_t ws_size,
                              hipStream_t stream) {
  (void)in_sizes; (void)n_in; (void)out_size; (void)ws_size;
  const float* q  = (const float*)d_in[0];
  const float* k  = (const float*)d_in[1];
  const float* v  = (const float*)d_in[2];
  // d_in[3] = mask: causal triu, handled analytically in attn_flash
  const float* Wq = (const float*)d_in[4];
  const float* bq = (const float*)d_in[5];
  const float* Wk = (const float*)d_in[6];
  const float* bk = (const float*)d_in[7];
  const float* Wv = (const float*)d_in[8];
  const float* bv = (const float*)d_in[9];
  const float* Wo = (const float*)d_in[10];
  const float* bo = (const float*)d_in[11];
  float* out = (float*)d_out;

  char* ws = (char*)d_ws;
  const size_t MB = 1024 * 1024;
  _Float16* q16  = (_Float16*)(ws + 0 * MB);
  _Float16* k16  = (_Float16*)(ws + 8 * MB);
  _Float16* v16  = (_Float16*)(ws + 16 * MB);
  _Float16* wq16 = (_Float16*)(ws + 24 * MB);
  _Float16* wk16 = (_Float16*)(ws + 26 * MB);
  _Float16* wv16 = (_Float16*)(ws + 28 * MB);
  _Float16* wo16 = (_Float16*)(ws + 30 * MB);
  _Float16* Qh   = (_Float16*)(ws + 32 * MB);   // [B,H,S,64] (pre-scaled)
  _Float16* Kh   = (_Float16*)(ws + 40 * MB);   // [B,H,S,64]
  _Float16* Vt   = (_Float16*)(ws + 48 * MB);   // [B,H,64,S] transposed
  _Float16* c16  = (_Float16*)(ws + 56 * MB);   // [B,S,1024]

  const int nX4 = MTOT * DMODEL / 4;    // 1,048,576
  const int nW4 = DMODEL * DMODEL / 4;  //   262,144
  cvt_f32_to_f16<<<nX4 / 256, 256, 0, stream>>>(q, q16, nX4);
  cvt_f32_to_f16<<<nX4 / 256, 256, 0, stream>>>(k, k16, nX4);
  cvt_f32_to_f16<<<nX4 / 256, 256, 0, stream>>>(v, v16, nX4);
  cvt_f32_to_f16<<<nW4 / 256, 256, 0, stream>>>(Wq, wq16, nW4);
  cvt_f32_to_f16<<<nW4 / 256, 256, 0, stream>>>(Wk, wk16, nW4);
  cvt_f32_to_f16<<<nW4 / 256, 256, 0, stream>>>(Wv, wv16, nW4);
  cvt_f32_to_f16<<<nW4 / 256, 256, 0, stream>>>(Wo, wo16, nW4);

  // 2048 tiles (128 Mx16 N) = 2048 waves = 256 blocks of 8 waves; z selects Q/K/V
  proj_qkv<<<dim3(256, 1, 3), 256, 0, stream>>>(q16, k16, v16, wq16, wk16, wv16,
                                                bq, bk, bv, Qh, Kh, Vt);

  // B*H*(S/16) = 4096 waves = 512 blocks
  attn_flash<<<512, 256, 0, stream>>>(Qh, Kh, Vt, c16);

  proj_out<<<256, 256, 0, stream>>>(c16, wo16, bo, out);
}
